// RnnNet_19688130085224
// MI455X (gfx1250) — matrix operand order repaired and checked by reading, not compiled, not run on hardware
//
#include <hip/hip_runtime.h>
#include <hip/hip_bf16.h>
#include <math.h>

// ---------------------------------------------------------------------------
// Fused GRU(9->5) x48 + BatchNorm(T) + Linear(240->1) + sigmoid, MI455X.
// Phase A: per-wave 16-batch tile; recurrence on v_wmma_f32_16x16x32_f16.
//          Cross-lane moves use v_permlane(x)16 (VALU) instead of LDS.
//          Gate nonlinearities use v_tanh_f32 when available.
// Phase B: per-t BN coefficients + scalar offset.
// Phase C: out = sigmoid(sum_t coef[t]*p[b,t] + offset), p stored as f16.
// ---------------------------------------------------------------------------

typedef __attribute__((ext_vector_type(16))) _Float16 v16h;
typedef __attribute__((ext_vector_type(8)))  float    v8f;
typedef __attribute__((ext_vector_type(2)))  __fp16   fp16x2;

#define BATCH 262144
#define TT    48
#define FF    9
#define HH    5

__device__ __forceinline__ float fexp2(float x) {
#if __has_builtin(__builtin_amdgcn_exp2f)
  return __builtin_amdgcn_exp2f(x);
#else
  return exp2f(x);
#endif
}
__device__ __forceinline__ float frcp(float x) {
#if __has_builtin(__builtin_amdgcn_rcpf)
  return __builtin_amdgcn_rcpf(x);
#else
  return 1.0f / x;
#endif
}
__device__ __forceinline__ float ftanh_(float x) {
#if __has_builtin(__builtin_amdgcn_tanhf)
  return __builtin_amdgcn_tanhf(x);            // v_tanh_f32 (CDNA5 TRANS op)
#else
  return 2.0f * frcp(1.0f + fexp2(-2.88539008f * x)) - 1.0f;
#endif
}
__device__ __forceinline__ float fsig(float x) {  // sigmoid via tanh: 1 trans op
#if __has_builtin(__builtin_amdgcn_tanhf)
  return __builtin_fmaf(0.5f, __builtin_amdgcn_tanhf(0.5f * x), 0.5f);
#else
  return frcp(1.0f + fexp2(-1.44269504f * x));
#endif
}

// f32 pair -> packed f16x2 dword (v_cvt_pk_f16_f32 rtz)
__device__ __forceinline__ unsigned pack2h(float a, float b) {
  fp16x2 p = __builtin_amdgcn_cvt_pkrtz(a, b);
  return __builtin_bit_cast(unsigned, p);
}

// lane^16 half-wave swap as pure VALU (v_permlanex16_b32, identity selects)
__device__ __forceinline__ unsigned swap16u(unsigned v) {
#if __has_builtin(__builtin_amdgcn_permlanex16)
  return __builtin_amdgcn_permlanex16(v, v, 0x76543210u, 0xFEDCBA98u, false, false);
#else
  return __shfl_xor((int)v, 16, 32);
#endif
}
__device__ __forceinline__ float swap16(float v) {
  return __builtin_bit_cast(float, swap16u(__builtin_bit_cast(unsigned, v)));
}

__global__ void zero_stats(float* ws) {
  int i = threadIdx.x;
  if (i < 160) ws[i] = 0.0f;
}

// Gate-row remap (PyTorch gate rows: r=0-4, z=5-9, n=10-14):
// tile rows 0-4  -> r0-4  (lower lanes, D v0-4)
// tile rows 5,6  -> z0,z1 (lower lanes, D v5,v6)
// tile row  7    -> zero pad
// tile rows 8-12 -> n0-4  (upper lanes, D v0-4)
// tile rows 13-15-> z2,z3,z4 (upper lanes, D v5,v6,v7)
__device__ __forceinline__ int orig_row(int tr) {
  if (tr < 7)  return tr;        // r0-4, z0, z1
  if (tr == 7) return -1;        // padding
  if (tr < 13) return tr + 2;    // n0-4 -> rows 10-14
  return tr - 6;                 // 13,14,15 -> 7,8,9 (z2,z3,z4)
}

__global__ __launch_bounds__(256) void gru_phaseA(
    const float* __restrict__ x,    const float* __restrict__ Wih,
    const float* __restrict__ Whh,  const float* __restrict__ bih,
    const float* __restrict__ bhh,  const float* __restrict__ Wlin,
    _Float16* __restrict__ pbuf,    float* __restrict__ gstats) {
  // per-(t, lane) stat partials: [0]=sum, [1]=sumsq; 16 lanes -> 16 banks
  __shared__ float lstat[2][TT][16];

  const int tid  = threadIdx.x;
  const int lane = tid & 31;
  const int wave = tid >> 5;
  for (int i = tid; i < 2 * TT * 16; i += 256)
    (&lstat[0][0][0])[i] = 0.0f;
  __syncthreads();

  const int tileBase = (blockIdx.x * 8 + wave) * 16;

  // ---- A operands (wave-constant f16 weights, gate rows remapped) ----
  // A 16x32 f16: lane m(<16) = K0-7 in elems0-7; lane m+16 = K24,K25 (f8, bias) in elems 8,9;
  // all other elems stay zero -> matching B slots are don't-care.
  union { v16h v; _Float16 h[16]; unsigned u[8]; } ax, ah, bx, bh;
  #pragma unroll
  for (int e = 0; e < 16; ++e) { ax.h[e] = (_Float16)0.f; ah.h[e] = (_Float16)0.f;
                                 bx.h[e] = (_Float16)0.f; bh.h[e] = (_Float16)0.f; }
  {
    const int m = lane & 15;
    const int g = orig_row(m);
    if (g >= 0) {
      if (lane < 16) {
        #pragma unroll
        for (int e = 0; e < 8; ++e) ax.h[e] = (_Float16)Wih[g * FF + e];  // K0-7
        #pragma unroll
        for (int e = 0; e < HH; ++e) ah.h[e] = (_Float16)Whh[g * HH + e]; // K0-4
        ah.h[5] = (_Float16)bhh[g];                                       // K5*1
      } else {
        ax.h[8] = (_Float16)Wih[g * FF + 8];                              // K24=f8
        ax.h[9] = (_Float16)bih[g];                                       // K25*1
      }
    }
  }
  bx.h[9] = (_Float16)1.0f;  // K25 constant-1 in upper lanes (input bias)
  bh.h[5] = (_Float16)1.0f;  // K5 constant-1 (hidden bias)

  // h state fp32, valid in UPPER lanes (lane 16+b)
  float hup0 = 0.f, hup1 = 0.f, hup2 = 0.f, hup3 = 0.f, hup4 = 0.f;

  const int bl = lane & 15;  // batch index within tile (both halves mirror it)
  const float* __restrict__ xrow = x + (size_t)(tileBase + bl) * (TT * FF);

  #pragma unroll 2
  for (int t = 0; t < TT; ++t) {
    // x[b,t,0..8] -> Bx elems 0-8 (all lanes load; upper-lane dups coalesce;
    // only x0-7 at K0-7 (lower) and x8 at K24 (upper) meet nonzero A)
    {
      const float* xr = xrow + t * FF;
      #pragma unroll
      for (int f = 0; f < FF; ++f) bx.h[f] = (_Float16)xr[f];
    }

    v8f zero = {};
    v8f Dx = __builtin_amdgcn_wmma_f32_16x16x32_f16(   // [Wih|bih]*[x;1]
        false, ax.v, false, bx.v, (short)0, zero, false, false);
    v8f Dh = __builtin_amdgcn_wmma_f32_16x16x32_f16(   // [Whh|bhh]*[h;1]
        false, ah.v, false, bh.v, (short)0, zero, false, false);

    // sigmoid slots: r (lower v0-4), z0,z1 (lower v5,v6), z2,z3,z4 (upper v5-7)
    float s0 = fsig(Dx[0] + Dh[0]);
    float s1 = fsig(Dx[1] + Dh[1]);
    float s2 = fsig(Dx[2] + Dh[2]);
    float s3 = fsig(Dx[3] + Dh[3]);
    float s4 = fsig(Dx[4] + Dh[4]);
    float s5 = fsig(Dx[5] + Dh[5]);
    float s6 = fsig(Dx[6] + Dh[6]);
    float s7 = fsig(Dx[7] + Dh[7]);

    // r -> upper lanes (where the n rows live)
    float r0 = swap16(s0), r1 = swap16(s1), r2 = swap16(s2),
          r3 = swap16(s3), r4 = swap16(s4);
    // n = tanh(xn + r*hn) (valid upper; Dx/Dh v0-4 upper = n rows)
    float n0 = ftanh_(Dx[0] + r0 * Dh[0]);
    float n1 = ftanh_(Dx[1] + r1 * Dh[1]);
    float n2 = ftanh_(Dx[2] + r2 * Dh[2]);
    float n3 = ftanh_(Dx[3] + r3 * Dh[3]);
    float n4 = ftanh_(Dx[4] + r4 * Dh[4]);
    // z -> upper lanes (z2..z4 are already there)
    float z0 = swap16(s5), z1 = swap16(s6);
    float z2 = s5, z3 = s6, z4 = s7;
    // h' = n + z*(h - n)
    hup0 = n0 + z0 * (hup0 - n0);
    hup1 = n1 + z1 * (hup1 - n1);
    hup2 = n2 + z2 * (hup2 - n2);
    hup3 = n3 + z3 * (hup3 - n3);
    hup4 = n4 + z4 * (hup4 - n4);

    // per-t stats + partial linear dot (valid upper)
    float s  = hup0 + hup1 + hup2 + hup3 + hup4;
    float ss = hup0*hup0 + hup1*hup1 + hup2*hup2 + hup3*hup3 + hup4*hup4;
    float p  = hup0 * Wlin[t*HH+0] + hup1 * Wlin[t*HH+1] + hup2 * Wlin[t*HH+2]
             + hup3 * Wlin[t*HH+3] + hup4 * Wlin[t*HH+4];
    if (lane >= 16) {
      // 16 lanes -> 16 distinct banks: one vector ds_add each, no reduce tree
      atomicAdd(&lstat[0][t][lane - 16], s);
      atomicAdd(&lstat[1][t][lane - 16], ss);
      pbuf[(size_t)t * BATCH + tileBase + (lane - 16)] = (_Float16)p;
    }

    // rebuild Bh: pack f16 pairs in upper lanes, swap 3 packed dwords down
    bh.u[0] = swap16u(pack2h(hup0, hup1));  // K0,K1 = h0,h1
    bh.u[1] = swap16u(pack2h(hup2, hup3));  // K2,K3 = h2,h3
    bh.u[2] = swap16u(pack2h(hup4, 1.0f));  // K4,K5 = h4,1 (bias slot)
  }

  __syncthreads();
  // fold 16 lane-partials per (stat, t) and push to global accumulators
  if (tid < 2 * TT) {
    const float* row = &lstat[tid / TT][tid % TT][0];
    float acc = 0.0f;
    #pragma unroll
    for (int i = 0; i < 16; ++i) acc += row[i];
    atomicAdd(&gstats[tid], acc);
  }
}

__global__ void gru_phaseB(const float* __restrict__ gamma,
                           const float* __restrict__ beta,
                           const float* __restrict__ Wlin,
                           const float* __restrict__ blin,
                           float* __restrict__ ws) {
  __shared__ float red[64];
  int t = threadIdx.x;
  float po = 0.0f;
  if (t < TT) {
    const float N  = (float)BATCH * (float)HH;
    float mean = ws[t] / N;
    float var  = ws[TT + t] / N - mean * mean;
    float a    = gamma[t] * rsqrtf(var + 1e-5f);
    float c    = beta[t] - mean * a;
    ws[96 + t] = a;
    float wsum = 0.0f;
    #pragma unroll
    for (int h = 0; h < HH; ++h) wsum += Wlin[t * HH + h];
    po = c * wsum;
  }
  red[t] = po;
  __syncthreads();
  if (t == 0) {
    float acc = blin[0];
    #pragma unroll
    for (int i = 0; i < 64; ++i) acc += red[i];
    ws[144] = acc;
  }
}

__global__ __launch_bounds__(256) void gru_phaseC(
    const float* __restrict__ ws, const _Float16* __restrict__ pbuf,
    float* __restrict__ out) {
  int b = blockIdx.x * 256 + threadIdx.x;
  float acc = ws[144];
  #pragma unroll
  for (int t = 0; t < TT; ++t)
    acc += ws[96 + t] * (float)pbuf[(size_t)t * BATCH + b];
  out[b] = fsig(acc);
}

extern "C" void kernel_launch(void* const* d_in, const int* in_sizes, int n_in,
                              void* d_out, int out_size, void* d_ws, size_t ws_size,
                              hipStream_t stream) {
  const float* x     = (const float*)d_in[0];
  const float* W_ih  = (const float*)d_in[1];
  const float* W_hh  = (const float*)d_in[2];
  const float* b_ih  = (const float*)d_in[3];
  const float* b_hh  = (const float*)d_in[4];
  const float* gamma = (const float*)d_in[5];
  const float* beta  = (const float*)d_in[6];
  const float* W_lin = (const float*)d_in[7];
  const float* b_lin = (const float*)d_in[8];
  float* out = (float*)d_out;

  // ws floats: [0..47]=sum, [48..95]=sumsq, [96..143]=coef, [144]=offset
  // p[T,B] as f16 starts at float offset 256 (~24 MB of scratch)
  float*    ws   = (float*)d_ws;
  _Float16* pbuf = (_Float16*)(ws + 256);

  zero_stats<<<1, 256, 0, stream>>>(ws);
  gru_phaseA<<<BATCH / 128, 256, 0, stream>>>(x, W_ih, W_hh, b_ih, b_hh,
                                              W_lin, pbuf, ws);
  gru_phaseB<<<1, 64, 0, stream>>>(gamma, beta, W_lin, b_lin, ws);
  gru_phaseC<<<BATCH / 256, 256, 0, stream>>>(ws, pbuf, out);
}
